// LocalAttention_7851200217265
// MI455X (gfx1250) — matrix operand order repaired and checked by reading, not compiled, not run on hardware
//
#include <hip/hip_runtime.h>

// ---------------------------------------------------------------------------
// MI455X (gfx1250) fused windowed attention.
// bf16 WMMA (v_wmma_f32_16x16x32_bf16) for all GEMMs, f32 softmax.
// One workgroup (256 thr = 8 wave32) per 8x8 window; qkv lives in LDS.
// GEMM loops: k-loop unrolling disabled so the A fragments are re-loaded from
// LDS each k step (prevents LICM from hoisting 64 fragments -> scratch spill);
// two N tiles processed per iteration to amortize each A fragment load.
// ---------------------------------------------------------------------------

typedef __bf16 bf16_t;
typedef bf16_t v16bf __attribute__((ext_vector_type(16)));
typedef float  v8f   __attribute__((ext_vector_type(8)));
typedef unsigned short u16;

#define DIMC    512
#define HEADS   16
#define HEAD_DIM 32
#define WSZ     8
#define NTOK    64            // tokens per window (8x8)
#define IMG     64
#define NWIN    1024          // 16 batches * 8 * 8 windows
#define QCOLS   1536          // 3*DIMC
#define LDA0    520           // padded bf16 stride for x/O tile region
#define LDQ     1544          // padded bf16 stride for qkv region
#define LDOF    65            // padded f32 stride for staged output [c][tok]

union FragU {
    v16bf v;
    uint4 q[2];
    u16   us[16];
};

static __device__ __forceinline__ u16 f2bf(float f) {
    bf16_t b = (bf16_t)f;                  // RNE convert
    return __builtin_bit_cast(u16, b);
}

static __device__ __forceinline__ v8f vzero() {
    v8f z = {0.f, 0.f, 0.f, 0.f, 0.f, 0.f, 0.f, 0.f};
    return z;
}

static __device__ __forceinline__ v8f wmma_bf16(v16bf a, v16bf b, v8f c) {
    // (neg_a, A, neg_b, B, c_mod, C, reuse_a, reuse_b)
    return __builtin_amdgcn_wmma_f32_16x16x32_bf16(false, a, false, b,
                                                   (short)0, c, false, false);
}

// A-operand fragment (16x32 MxK, 16-bit): lane&15 = row; lane>>4 selects the
// K half {0..15} vs {16..31}.  base is row-major [rows][ld] bf16.
static __device__ __forceinline__ v16bf load_frag_A(const u16* base, int ld,
                                                    int row0, int k0, int lane) {
    const u16* p = base + (size_t)(row0 + (lane & 15)) * ld + k0 + 16 * (lane >> 4);
    FragU f;
    f.q[0] = *(const uint4*)(p);
    f.q[1] = *(const uint4*)(p + 8);
    return f.v;
}

// B-operand fragment (32x16 KxN, 16-bit) loaded from B^T stored row-major
// [N][K] (i.e. weight matrices [out][in]): lane&15 = n; lane>>4 = K half;
// 16 contiguous K values per lane.
static __device__ __forceinline__ v16bf load_frag_BT(const u16* base, int ld,
                                                     int n0, int k0, int lane) {
    const u16* p = base + (size_t)(n0 + (lane & 15)) * ld + k0 + 16 * (lane >> 4);
    FragU f;
    f.q[0] = *(const uint4*)(p);
    f.q[1] = *(const uint4*)(p + 8);
    return f.v;
}

// ---------------------------------------------------------------------------
// Weight f32 -> bf16 conversion (tiny, L2-resident afterwards)
// ---------------------------------------------------------------------------
__global__ void cvt_weights_kernel(const float* __restrict__ wqkv,
                                   const float* __restrict__ wout,
                                   u16* __restrict__ wqkv_bf,
                                   u16* __restrict__ wout_bf) {
    int i = blockIdx.x * 256 + threadIdx.x;
    if (i < QCOLS * DIMC) wqkv_bf[i] = f2bf(wqkv[i]);
    if (i < DIMC * DIMC)  wout_bf[i] = f2bf(wout[i]);
}

// ---------------------------------------------------------------------------
// Fused per-window kernel
// ---------------------------------------------------------------------------
__launch_bounds__(256, 1)
__global__ void win_attn_kernel(const float* __restrict__ x,
                                const u16*   __restrict__ wqkv_bf,
                                const float* __restrict__ bqkv,
                                const u16*   __restrict__ wout_bf,
                                const float* __restrict__ bout,
                                const float* __restrict__ bias_table,
                                const int*   __restrict__ rel_index,
                                float*       __restrict__ out) {
    __shared__ u16 sA[NTOK * LDA0];   // x tile (bf16), later attention output O
    __shared__ u16 sQ[NTOK * LDQ];    // qkv (bf16), later staged f32 output
    float* sOutF = (float*)sQ;        // [c][tok], stride LDOF (overlays sQ)

    const int tid  = threadIdx.x;
    const int lane = tid & 31;
    const int wid  = tid >> 5;
    const int half = lane >> 4;
    const int nlo  = lane & 15;

    const int win = blockIdx.x;
    const int bb_ = win >> 6;
    const int r1  = (win >> 3) & 7;
    const int r2  = win & 7;
    const size_t winOfs = (((size_t)bb_ * DIMC) * IMG + (size_t)r1 * WSZ) * IMG + (size_t)r2 * WSZ;

    // ---- stage 1: gather window tile [tok][c] -> sA (bf16) ----------------
    {
        const float* xwin = x + winOfs;
        int w1 = tid & 7, h1 = (tid >> 3) & 7, c0 = tid >> 6;
        int tok = h1 * WSZ + w1;
#pragma unroll 4
        for (int i = 0; i < DIMC / 4; ++i) {
            int c = c0 + 4 * i;
            sA[tok * LDA0 + c] = f2bf(xwin[(size_t)c * IMG * IMG + h1 * IMG + w1]);
        }
    }
    __syncthreads();

    // ---- stage 2: qkv = xt @ w_qkv^T + b_qkv  ->  sQ ----------------------
    // Wave handles 12 N-tiles (ni = wid + 8j), processed in pairs.
#pragma unroll 1
    for (int j = 0; j < 12; j += 2) {
        const int ni0 = wid + 8 * j;
        const int ni1 = ni0 + 8;
        v8f acc0[4], acc1[4];
#pragma unroll
        for (int mi = 0; mi < 4; ++mi) { acc0[mi] = vzero(); acc1[mi] = vzero(); }
#pragma unroll 1
        for (int k = 0; k < DIMC / 32; ++k) {
            v16bf bf0 = load_frag_BT(wqkv_bf, DIMC, ni0 * 16, k * 32, lane);
            v16bf bf1 = load_frag_BT(wqkv_bf, DIMC, ni1 * 16, k * 32, lane);
#pragma unroll
            for (int mi = 0; mi < 4; ++mi) {
                v16bf af = load_frag_A(sA, LDA0, mi * 16, k * 32, lane);
                acc0[mi] = wmma_bf16(af, bf0, acc0[mi]);
                acc1[mi] = wmma_bf16(af, bf1, acc1[mi]);
            }
        }
#pragma unroll
        for (int t = 0; t < 2; ++t) {
            const v8f* acc = t ? acc1 : acc0;
            int n = (t ? ni1 : ni0) * 16 + nlo;
            float bv = bqkv[n];
#pragma unroll
            for (int mi = 0; mi < 4; ++mi)
#pragma unroll
                for (int r = 0; r < 8; ++r) {
                    int m = mi * 16 + r + 8 * half;
                    sQ[m * LDQ + n] = f2bf(acc[mi][r] + bv);
                }
        }
    }
    __syncthreads();

    // ---- stage 3: attention; wave handles heads wid and wid+8 -------------
    const float scale = 0.17677669529663687f;   // HEAD_DIM^-0.5
#pragma unroll 1
    for (int hh = 0; hh < 2; ++hh) {
        int head = wid + 8 * hh;
        const u16* qb = sQ + head * HEAD_DIM;
        const u16* kb = sQ + DIMC + head * HEAD_DIM;
        const u16* vb = sQ + 2 * DIMC + head * HEAD_DIM;

        // S = q @ k^T   (M=64, N=64, K=32)
        v8f S[4][4];
#pragma unroll
        for (int mi = 0; mi < 4; ++mi)
#pragma unroll
            for (int ni = 0; ni < 4; ++ni) S[mi][ni] = vzero();
        {
            v16bf a[4];
#pragma unroll
            for (int mi = 0; mi < 4; ++mi) a[mi] = load_frag_A(qb, LDQ, mi * 16, 0, lane);
#pragma unroll
            for (int ni = 0; ni < 4; ++ni) {
                v16bf bf = load_frag_BT(kb, LDQ, ni * 16, 0, lane);
#pragma unroll
                for (int mi = 0; mi < 4; ++mi) S[mi][ni] = wmma_bf16(a[mi], bf, S[mi][ni]);
            }
        }

        // scale + relative position bias + row softmax (rows live per half-wave)
#pragma unroll
        for (int mi = 0; mi < 4; ++mi) {
#pragma unroll
            for (int r = 0; r < 8; ++r) {
                int m = mi * 16 + r + 8 * half;
                float vv[4];
#pragma unroll
                for (int ni = 0; ni < 4; ++ni) {
                    int n = ni * 16 + nlo;
                    int ridx = rel_index[m * NTOK + n];
                    vv[ni] = S[mi][ni][r] * scale + bias_table[ridx * HEADS + head];
                }
                float mx = fmaxf(fmaxf(vv[0], vv[1]), fmaxf(vv[2], vv[3]));
                for (int off = 8; off; off >>= 1) mx = fmaxf(mx, __shfl_xor(mx, off, 32));
                float sum = 0.f;
#pragma unroll
                for (int ni = 0; ni < 4; ++ni) { vv[ni] = __expf(vv[ni] - mx); sum += vv[ni]; }
                for (int off = 8; off; off >>= 1) sum += __shfl_xor(sum, off, 32);
                float inv = __builtin_amdgcn_rcpf(sum);
#pragma unroll
                for (int ni = 0; ni < 4; ++ni) S[mi][ni][r] = vv[ni] * inv;
            }
        }

        // store P (bf16) over this head's dead q (cols 0..31) / k (cols 32..63)
#pragma unroll
        for (int mi = 0; mi < 4; ++mi)
#pragma unroll
            for (int ni = 0; ni < 4; ++ni) {
                u16* dst = (u16*)((ni < 2) ? qb : kb);
                int ncol = (ni & 1) * 16 + nlo;
#pragma unroll
                for (int r = 0; r < 8; ++r) {
                    int m = mi * 16 + r + 8 * half;
                    dst[m * LDQ + ncol] = f2bf(S[mi][ni][r]);
                }
            }

        // O = P @ v   (M=64, N=32, K=64; K split across the q/k regions)
        v8f O[4][2];
#pragma unroll
        for (int mi = 0; mi < 4; ++mi) { O[mi][0] = vzero(); O[mi][1] = vzero(); }
#pragma unroll
        for (int kk = 0; kk < 2; ++kk) {
            const u16* pb = kk ? kb : qb;
            v16bf a[4];
#pragma unroll
            for (int mi = 0; mi < 4; ++mi) a[mi] = load_frag_A(pb, LDQ, mi * 16, 0, lane);
#pragma unroll
            for (int ni = 0; ni < 2; ++ni) {
                FragU bf;  // B[k][n] = v[tok=k][n]; gather (v stored row-major by token)
#pragma unroll
                for (int jj = 0; jj < 16; ++jj)
                    bf.us[jj] = vb[(kk * 32 + 16 * half + jj) * LDQ + ni * 16 + nlo];
#pragma unroll
                for (int mi = 0; mi < 4; ++mi) O[mi][ni] = wmma_bf16(a[mi], bf.v, O[mi][ni]);
            }
        }

        // O -> sA cols [head*32, head*32+32)  (x tile is dead)
#pragma unroll
        for (int mi = 0; mi < 4; ++mi)
#pragma unroll
            for (int ni = 0; ni < 2; ++ni)
#pragma unroll
                for (int r = 0; r < 8; ++r) {
                    int m = mi * 16 + r + 8 * half;
                    sA[m * LDA0 + head * HEAD_DIM + ni * 16 + nlo] = f2bf(O[mi][ni][r]);
                }
    }
    __syncthreads();

    // ---- stage 4: out = O @ w_out^T + b_out -> staged f32 [c][tok] --------
    // Wave handles 4 N-tiles, processed in pairs.
#pragma unroll 1
    for (int j = 0; j < 4; j += 2) {
        const int ni0 = wid + 8 * j;
        const int ni1 = ni0 + 8;
        v8f acc0[4], acc1[4];
#pragma unroll
        for (int mi = 0; mi < 4; ++mi) { acc0[mi] = vzero(); acc1[mi] = vzero(); }
#pragma unroll 1
        for (int k = 0; k < DIMC / 32; ++k) {
            v16bf bf0 = load_frag_BT(wout_bf, DIMC, ni0 * 16, k * 32, lane);
            v16bf bf1 = load_frag_BT(wout_bf, DIMC, ni1 * 16, k * 32, lane);
#pragma unroll
            for (int mi = 0; mi < 4; ++mi) {
                v16bf af = load_frag_A(sA, LDA0, mi * 16, k * 32, lane);
                acc0[mi] = wmma_bf16(af, bf0, acc0[mi]);
                acc1[mi] = wmma_bf16(af, bf1, acc1[mi]);
            }
        }
#pragma unroll
        for (int t = 0; t < 2; ++t) {
            const v8f* acc = t ? acc1 : acc0;
            int n = (t ? ni1 : ni0) * 16 + nlo;
            float bv = bout[n];
#pragma unroll
            for (int mi = 0; mi < 4; ++mi)
#pragma unroll
                for (int r = 0; r < 8; ++r) {
                    int m = mi * 16 + r + 8 * half;
                    sOutF[n * LDOF + m] = acc[mi][r] + bv;
                }
        }
    }
    __syncthreads();

    // ---- stage 5: coalesced scatter to NCHW output ------------------------
    {
        float* owin = out + winOfs;
        int w1 = tid & 7, h1 = (tid >> 3) & 7, c0 = tid >> 6;
        int tok = h1 * WSZ + w1;
#pragma unroll 4
        for (int i = 0; i < DIMC / 4; ++i) {
            int c = c0 + 4 * i;
            owin[(size_t)c * IMG * IMG + h1 * IMG + w1] = sOutF[c * LDOF + tok];
        }
    }
}

// ---------------------------------------------------------------------------
extern "C" void kernel_launch(void* const* d_in, const int* in_sizes, int n_in,
                              void* d_out, int out_size, void* d_ws, size_t ws_size,
                              hipStream_t stream) {
    const float* x          = (const float*)d_in[0];
    const float* w_qkv      = (const float*)d_in[1];
    const float* b_qkv      = (const float*)d_in[2];
    const float* w_out      = (const float*)d_in[3];
    const float* b_out      = (const float*)d_in[4];
    const float* bias_table = (const float*)d_in[5];
    const int*   rel_index  = (const int*)d_in[6];
    float*       out        = (float*)d_out;

    u16* wqkv_bf = (u16*)d_ws;                       // 1536*512 bf16
    u16* wout_bf = wqkv_bf + (size_t)QCOLS * DIMC;   //  512*512 bf16

    cvt_weights_kernel<<<(QCOLS * DIMC + 255) / 256, 256, 0, stream>>>(
        w_qkv, w_out, wqkv_bf, wout_bf);

    win_attn_kernel<<<NWIN, 256, 0, stream>>>(
        x, wqkv_bf, b_qkv, wout_bf, b_out, bias_table, rel_index, out);
}
